// Attention_89739046683285
// MI455X (gfx1250) — matrix operand order repaired and checked by reading, not compiled, not run on hardware
//
#include <hip/hip_runtime.h>
#include <hip/hip_bf16.h>

typedef __bf16 bf16;
typedef __attribute__((ext_vector_type(16))) __bf16 v16bf;
typedef __attribute__((ext_vector_type(8)))  __bf16 v8bf;
typedef __attribute__((ext_vector_type(8)))  float  v8f;

#define NUMH 12
#define DH   64
#define SEQ  2048
#define BATCH 2
#define CDIM 768

union PkBf { bf16 b[2]; unsigned int u; };

static __device__ __forceinline__ unsigned int pack2(float a, float b) {
  PkBf p;
  p.b[0] = (bf16)a;
  p.b[1] = (bf16)b;
  return p.u;
}

static __device__ __forceinline__ v16bf make_v16(v8bf lo, v8bf hi) {
  return __builtin_shufflevector(lo, hi, 0,1,2,3,4,5,6,7,8,9,10,11,12,13,14,15);
}

static __device__ __forceinline__ v8f wmma_bf16(v16bf a, v16bf b, v8f c) {
  // D = A(16x32 bf16) * B(32x16 bf16) + C(16x16 f32)
  return __builtin_amdgcn_wmma_f32_16x16x32_bf16(false, a, false, b, (short)0, c, false, false);
}

// ---------------------------------------------------------------------------
// Generic bf16-WMMA GEMM:  D[M,N] = A[M,K](f32) * W[N,K]^T(f32) + bias[N]
// MODE 0: write f32 to outF (row stride Nout)
// MODE 1: scatter QKV columns into q/k/v bf16 buffers laid out (B,H,N,Dh)
// Block tile 64(M) x 128(N), BK=32, 256 threads = 8 waves (4 in M x 2 in N),
// each wave produces a 16x64 tile (4 WMMA accumulators).
// ---------------------------------------------------------------------------
template<int MODE>
__global__ __launch_bounds__(256)
void gemm_wmma(const float* __restrict__ A, const float* __restrict__ W,
               const float* __restrict__ bias, float* __restrict__ outF,
               bf16* __restrict__ qb, bf16* __restrict__ kb, bf16* __restrict__ vb,
               int M, int K, int Nout)
{
  __shared__ __align__(32) bf16 ldsA[64 * 32];   // [m][k]
  __shared__ __align__(32) bf16 ldsB[32 * 128];  // [k][n]
  const int tid  = threadIdx.x;
  const int lane = tid & 31;
  const int wid  = tid >> 5;
  const int wy   = wid >> 1;      // 0..3 : M sub-tile
  const int wx   = wid & 1;       // 0..1 : N sub-tile
  const int m15  = lane & 15;
  const int lh   = lane >> 4;     // lane half (A/K interleave)
  const int mbase = blockIdx.y * 64;
  const int nbase = blockIdx.x * 128;

  v8f acc[4] = {};

  const int nkt = K >> 5;
  for (int kt = 0; kt < nkt; ++kt) {
    // --- stage A tile (64x32): 8 f32 -> 4 packed dwords -> one b128 store
    {
      int am = tid >> 2;
      int ak = (tid & 3) * 8;
      const float* src = A + (size_t)(mbase + am) * K + kt * 32 + ak;
      uint4 v;
      v.x = pack2(src[0], src[1]);
      v.y = pack2(src[2], src[3]);
      v.z = pack2(src[4], src[5]);
      v.w = pack2(src[6], src[7]);
      *(uint4*)(ldsA + am * 32 + ak) = v;
    }
    // --- stage W tile (128x32) transposed to [k][n]; pack adjacent n pair
    {
      int np  = tid & 63;        // n pair -> n = 2*np, 2*np+1
      int kb8 = (tid >> 6) * 8;  // 0,8,16,24
      const float* s0 = W + (size_t)(nbase + 2 * np) * K + kt * 32 + kb8;
      const float* s1 = s0 + K;
      #pragma unroll
      for (int i = 0; i < 8; ++i)
        *(unsigned int*)(ldsB + (kb8 + i) * 128 + 2 * np) = pack2(s0[i], s1[i]);
    }
    __syncthreads();

    // A fragment: row m15 of wave's 16-row strip; K chunks per ISA layout
    const bf16* ap = ldsA + (wy * 16 + m15) * 32;
    v8bf alo = *(const v8bf*)(ap + lh * 8);
    v8bf ahi = *(const v8bf*)(ap + 16 + lh * 8);
    v16bf afrag = make_v16(alo, ahi);

    const bf16* bp = ldsB + wx * 64 + m15;   // lane%16 == N index
    #pragma unroll
    for (int j = 0; j < 4; ++j) {
      v16bf bfrag;   // 16 K values, same K order as A
      #pragma unroll
      for (int e = 0; e < 16; ++e)
        bfrag[e] = bp[((e & 7) + 8 * lh + 16 * (e >> 3)) * 128 + j * 16];
      acc[j] = wmma_bf16(afrag, bfrag, acc[j]);
    }
    __syncthreads();
  }

  // --- epilogue (C/D layout: row = r + 8*lh, col = m15)
  #pragma unroll
  for (int j = 0; j < 4; ++j) {
    int c = nbase + wx * 64 + j * 16 + m15;
    float bv = bias[c];
    #pragma unroll
    for (int r = 0; r < 8; ++r) {
      int row = mbase + wy * 16 + r + 8 * lh;
      float val = acc[j][r] + bv;
      if (MODE == 0) {
        outF[(size_t)row * Nout + c] = val;
      } else {
        int three = c / CDIM;           // 0=q 1=k 2=v
        int h = (c >> 6) % NUMH;
        int d = c & 63;
        int b = row >> 11;              // row / SEQ
        int n = row & (SEQ - 1);
        size_t idx = (((size_t)b * NUMH + h) * SEQ + n) * DH + d;
        bf16 v = (bf16)val;
        if (three == 0)      qb[idx] = v;
        else if (three == 1) kb[idx] = v;
        else                 vb[idx] = v;
      }
    }
  }
}

// ---------------------------------------------------------------------------
// RoPE-2D on q and k in-place; q additionally scaled by Dh^-0.5.
// One thread per (b,h,n, axis, i in 0..15) rotation pair.
// ---------------------------------------------------------------------------
__global__ __launch_bounds__(256)
void rope_kernel(bf16* __restrict__ q, bf16* __restrict__ k,
                 const int* __restrict__ xpos)
{
  int t = blockIdx.x * blockDim.x + threadIdx.x;  // B*H*N*32 threads
  int i   = t & 15;
  int ax  = (t >> 4) & 1;         // 0: y-half (dims 0..31), 1: x-half (32..63)
  int bhn = t >> 5;
  int n   = bhn & (SEQ - 1);
  int bh  = bhn >> 11;
  int b   = bh / NUMH;

  int pos = xpos[((size_t)b * SEQ + n) * 2 + ax];
  float freq = (float)pos * __powf(100.0f, -(float)i / 16.0f);
  float s, c;
  __sincosf(freq, &s, &c);

  size_t base = ((size_t)bh * SEQ + n) * DH + ax * 32;
  const float qs = 0.125f;  // 1/sqrt(64)
  float q1 = (float)q[base + i], q2 = (float)q[base + i + 16];
  q[base + i]      = (bf16)((q1 * c - q2 * s) * qs);
  q[base + i + 16] = (bf16)((q2 * c + q1 * s) * qs);
  float k1 = (float)k[base + i], k2 = (float)k[base + i + 16];
  k[base + i]      = (bf16)(k1 * c - k2 * s);
  k[base + i + 16] = (bf16)(k2 * c + k1 * s);
}

// ---------------------------------------------------------------------------
// Flash attention: block = 4 waves = 64 q rows of one (b,h).
// Stream 64-key chunks. V chunk is DMA'd with GLOBAL_LOAD_ASYNC_TO_LDS_B128
// (ASYNCcnt) while K is transposed into LDS ([d][key], packed dword stores).
// Online softmax in registers; P re-laid out via per-wave LDS buffer.
// ---------------------------------------------------------------------------
__global__ __launch_bounds__(128)
void flash_attn(const bf16* __restrict__ qg, const bf16* __restrict__ kg,
                const bf16* __restrict__ vg, float* __restrict__ outp)
{
  __shared__ __align__(32) bf16 ldsKT[64 * 64];      // [d][key]
  __shared__ __align__(32) bf16 ldsV [64 * 64];      // [key][d]
  __shared__ __align__(32) bf16 ldsP [4 * 16 * 64];  // per-wave P tile

  const int tid  = threadIdx.x;
  const int lane = tid & 31;
  const int w    = tid >> 5;
  const int m15  = lane & 15;
  const int lh   = lane >> 4;
  const int bh   = blockIdx.y;           // 0..B*H-1
  const int h    = bh % NUMH;
  const int b    = bh / NUMH;
  const int qbase = blockIdx.x * 64;
  const size_t headoff = (size_t)bh * SEQ * DH;

  // Q fragments for this wave's 16 rows (Dh=64 -> two K=32 slices)
  const int qrow = qbase + w * 16 + m15;
  v16bf qf[2];
  #pragma unroll
  for (int s = 0; s < 2; ++s) {
    const bf16* qp = qg + headoff + (size_t)qrow * DH + s * 32;
    v8bf lo = *(const v8bf*)(qp + lh * 8);
    v8bf hi = *(const v8bf*)(qp + 16 + lh * 8);
    qf[s] = make_v16(lo, hi);
  }

  v8f oacc[4] = {};
  float mrow[8], lrow[8];
  #pragma unroll
  for (int r = 0; r < 8; ++r) { mrow[r] = -1e30f; lrow[r] = 0.0f; }

  bf16* myP = ldsP + w * (16 * 64);

  for (int kc = 0; kc < SEQ / 64; ++kc) {
    __syncthreads();   // previous chunk fully consumed

    // --- V chunk: async DMA straight into LDS (64B per thread, no VGPRs)
    {
      int key = tid >> 1;
      int db  = (tid & 1) * 32;
      const bf16* vs = vg + headoff + (size_t)(kc * 64 + key) * DH + db;
      unsigned int loff = (unsigned int)(uintptr_t)(ldsV + key * 64 + db);
      asm volatile(
        "global_load_async_to_lds_b128 %0, %1, off\n\t"
        "global_load_async_to_lds_b128 %0, %1, off offset:16\n\t"
        "global_load_async_to_lds_b128 %0, %1, off offset:32\n\t"
        "global_load_async_to_lds_b128 %0, %1, off offset:48"
        :: "v"(loff), "v"(vs) : "memory");
    }
    // --- K chunk transposed: pack adjacent key pair into one dword
    {
      int kp = tid & 31;           // keys 2*kp, 2*kp+1
      int db = (tid >> 5) * 16;    // 0,16,32,48
      const bf16* s0 = kg + headoff + (size_t)(kc * 64 + 2 * kp) * DH + db;
      const bf16* s1 = s0 + DH;
      #pragma unroll
      for (int i = 0; i < 16; ++i) {
        PkBf p;
        p.b[0] = s0[i];
        p.b[1] = s1[i];
        *(unsigned int*)(ldsKT + (db + i) * 64 + 2 * kp) = p.u;
      }
    }
    asm volatile("s_wait_asynccnt 0" ::: "memory");
    __syncthreads();

    // S = Q * K^T  (4 key-blocks of 16, two Dh slices each)
    v8f sacc[4] = {};
    #pragma unroll
    for (int j = 0; j < 4; ++j) {
      #pragma unroll
      for (int s = 0; s < 2; ++s) {
        const bf16* kp = ldsKT + (s * 32) * 64 + j * 16 + m15;
        v16bf bfrag;
        #pragma unroll
        for (int e = 0; e < 16; ++e)
          bfrag[e] = kp[((e & 7) + 8 * lh + 16 * (e >> 3)) * 64];
        sacc[j] = wmma_bf16(qf[s], bfrag, sacc[j]);
      }
    }

    // online softmax (rows of this half: r + 8*lh)
    #pragma unroll
    for (int r = 0; r < 8; ++r) {
      float mx = fmaxf(fmaxf(sacc[0][r], sacc[1][r]),
                       fmaxf(sacc[2][r], sacc[3][r]));
      #pragma unroll
      for (int off = 1; off < 16; off <<= 1)
        mx = fmaxf(mx, __shfl_xor(mx, off, 32));
      float mnew = fmaxf(mrow[r], mx);
      float corr = __expf(mrow[r] - mnew);
      mrow[r] = mnew;
      lrow[r] *= corr;
      #pragma unroll
      for (int j = 0; j < 4; ++j) oacc[j][r] *= corr;
      float psum = 0.0f;
      #pragma unroll
      for (int j = 0; j < 4; ++j) {
        float p = __expf(sacc[j][r] - mnew);
        psum += p;
        myP[(r + 8 * lh) * 64 + j * 16 + m15] = (bf16)p;
      }
      #pragma unroll
      for (int off = 1; off < 16; off <<= 1)
        psum += __shfl_xor(psum, off, 32);
      lrow[r] += psum;
    }
    asm volatile("s_wait_dscnt 0" ::: "memory");  // P visible to own wave

    // O += P * V (two 32-key slices, 4 d-blocks of 16)
    #pragma unroll
    for (int c = 0; c < 2; ++c) {
      const bf16* pp = myP + m15 * 64 + c * 32;
      v8bf plo = *(const v8bf*)(pp + lh * 8);
      v8bf phi = *(const v8bf*)(pp + 16 + lh * 8);
      v16bf pf = make_v16(plo, phi);
      #pragma unroll
      for (int j = 0; j < 4; ++j) {
        const bf16* vp = ldsV + (c * 32) * 64 + j * 16 + m15;
        v16bf vf;
        #pragma unroll
        for (int e = 0; e < 16; ++e)
          vf[e] = vp[((e & 7) + 8 * lh + 16 * (e >> 3)) * 64];
        oacc[j] = wmma_bf16(pf, vf, oacc[j]);
      }
    }
  }

  // write (B,N,C) fp32 for projection GEMM
  #pragma unroll
  for (int j = 0; j < 4; ++j) {
    #pragma unroll
    for (int r = 0; r < 8; ++r) {
      int qn = qbase + w * 16 + r + 8 * lh;
      outp[((size_t)b * SEQ + qn) * CDIM + h * DH + j * 16 + m15] =
          oacc[j][r] * (1.0f / lrow[r]);
    }
  }
}

// ---------------------------------------------------------------------------
extern "C" void kernel_launch(void* const* d_in, const int* in_sizes, int n_in,
                              void* d_out, int out_size, void* d_ws, size_t ws_size,
                              hipStream_t stream) {
  const float* x      = (const float*)d_in[0];
  const int*   xpos   = (const int*)  d_in[1];
  const float* w_qkv  = (const float*)d_in[2];
  const float* b_qkv  = (const float*)d_in[3];
  const float* w_proj = (const float*)d_in[4];
  const float* b_proj = (const float*)d_in[5];
  float* out = (float*)d_out;

  const size_t QE = (size_t)BATCH * NUMH * SEQ * DH;   // per-tensor elems
  bf16* qb = (bf16*)d_ws;
  bf16* kb = qb + QE;
  bf16* vb = kb + QE;
  float* attnout = (float*)(vb + QE);                  // (B,N,C) fp32

  const int M = BATCH * SEQ;                           // 4096

  // 1) QKV GEMM + scatter into (B,H,N,Dh) bf16 buffers
  gemm_wmma<1><<<dim3((3 * CDIM) / 128, M / 64), 256, 0, stream>>>(
      x, w_qkv, b_qkv, nullptr, qb, kb, vb, M, CDIM, 3 * CDIM);

  // 2) RoPE-2D (+ fold Dh^-0.5 into q)
  int total = BATCH * NUMH * SEQ * 32;
  rope_kernel<<<total / 256, 256, 0, stream>>>(qb, kb, xpos);

  // 3) Flash attention -> attnout (B,N,C) fp32
  flash_attn<<<dim3(SEQ / 64, BATCH * NUMH), 128, 0, stream>>>(
      qb, kb, vb, attnout);

  // 4) Output projection
  gemm_wmma<0><<<dim3(CDIM / 128, M / 64), 256, 0, stream>>>(
      attnout, w_proj, b_proj, out, nullptr, nullptr, nullptr, M, CDIM, CDIM);
}
